// SUTTransformer_42958262895125
// MI455X (gfx1250) — matrix operand order repaired and checked by reading, not compiled, not run on hardware
//
#include <hip/hip_runtime.h>

// ---------------------------------------------------------------------------
// CDNA5 (gfx1250) MoE transformer block, wave32 + v_wmma_f32_16x16x32_bf16.
// ---------------------------------------------------------------------------

typedef __attribute__((ext_vector_type(16))) __bf16 v16bf;
typedef __attribute__((ext_vector_type(8)))  float  v8f;

union Frag { v16bf v; unsigned int u[8]; };

__device__ inline unsigned short f2bf(float f) {
  unsigned int u = __float_as_uint(f);
  u += 0x7fffu + ((u >> 16) & 1u);          // round-to-nearest-even
  return (unsigned short)(u >> 16);
}
__device__ inline float bf2f(unsigned short h) {
  return __uint_as_float(((unsigned int)h) << 16);
}
__device__ inline v8f v8f_zero() {
  v8f z = {0.f, 0.f, 0.f, 0.f, 0.f, 0.f, 0.f, 0.f};
  return z;
}
__device__ inline v8f wmma_bf16(const Frag& a, const Frag& b, v8f c) {
  return __builtin_amdgcn_wmma_f32_16x16x32_bf16(false, a.v, false, b.v,
                                                 (short)0, c, false, false);
}

// ---------------------------------------------------------------------------
// elementwise converts
// ---------------------------------------------------------------------------
__global__ void to_bf16_kernel(const float* __restrict__ in,
                               unsigned short* __restrict__ out, long n) {
  long i = (long)blockIdx.x * blockDim.x + threadIdx.x;
  if (i < n) out[i] = f2bf(in[i]);
}

// rel_table [129,128] f32 -> relT [128][129] bf16
__global__ void transpose_rel_kernel(const float* __restrict__ in,
                                     unsigned short* __restrict__ out) {
  int i = blockIdx.x * blockDim.x + threadIdx.x;
  if (i < 128 * 129) {
    int h = i / 129, r = i % 129;
    out[i] = f2bf(in[r * 128 + h]);
  }
}

// ---------------------------------------------------------------------------
// LayerNorm: f32 row -> bf16 row
// ---------------------------------------------------------------------------
__device__ inline float block_reduce(float v, float* red, int tid) {
  red[tid] = v; __syncthreads();
  for (int st = 128; st > 0; st >>= 1) {
    if (tid < st) red[tid] += red[tid + st];
    __syncthreads();
  }
  float r = red[0]; __syncthreads();
  return r;
}

__global__ __launch_bounds__(256) void layernorm_bf16_kernel(
    const float* __restrict__ X, const float* __restrict__ w,
    const float* __restrict__ b, unsigned short* __restrict__ Out, int D) {
  __shared__ float red[256];
  int row = blockIdx.x, tid = threadIdx.x;
  const float* x = X + (size_t)row * D;
  float s = 0.f, s2 = 0.f;
  for (int d = tid; d < D; d += 256) { float v = x[d]; s += v; s2 += v * v; }
  float mean = block_reduce(s, red, tid) / (float)D;
  float var  = block_reduce(s2, red, tid) / (float)D - mean * mean;
  float inv  = rsqrtf(var + 1e-5f);
  for (int d = tid; d < D; d += 256) {
    float v = (x[d] - mean) * inv * w[d] + b[d];
    Out[(size_t)row * D + d] = f2bf(v);
  }
}

// ---------------------------------------------------------------------------
// Router: logits = xn @ Gw  -> top-2 softmax -> dense gates [T,E]
// ---------------------------------------------------------------------------
template <int E>
__global__ __launch_bounds__(256) void router_top2_kernel(
    const unsigned short* __restrict__ Xbf, const float* __restrict__ Gw,
    float* __restrict__ Gates, int D) {
  __shared__ float red[256];
  int row = blockIdx.x, tid = threadIdx.x;
  float part[E];
#pragma unroll
  for (int e = 0; e < E; ++e) part[e] = 0.f;
  for (int d = tid; d < D; d += 256) {
    float xv = bf2f(Xbf[(size_t)row * D + d]);
#pragma unroll
    for (int e = 0; e < E; ++e) part[e] += xv * Gw[(size_t)d * E + e];
  }
  float lg[E];
#pragma unroll
  for (int e = 0; e < E; ++e) lg[e] = block_reduce(part[e], red, tid);
  if (tid == 0) {
    int i1 = 0; float v1 = -1e30f;
    for (int e = 0; e < E; ++e) if (lg[e] > v1) { v1 = lg[e]; i1 = e; }
    int i2 = 0; float v2 = -1e30f;
    for (int e = 0; e < E; ++e) if (e != i1 && lg[e] > v2) { v2 = lg[e]; i2 = e; }
    float e1 = 1.f, e2 = __expf(v2 - v1);
    float inv = 1.f / (e1 + e2);
    for (int e = 0; e < E; ++e) Gates[(size_t)row * E + e] = 0.f;
    Gates[(size_t)row * E + i1] = e1 * inv;
    Gates[(size_t)row * E + i2] = e2 * inv;
  }
}

// ---------------------------------------------------------------------------
// Batched bf16 WMMA GEMM: C[z] = epilogue(A[z] @ B[z])
//   A row-major [M x K] (lda, + aBatch*z column/element offset)
//   B row-major [K x N] (ldb, + bBatch*z), C (ldc, + cBatch*z)
//   64x64 block tile, 8 waves, each wave: 32(M) x 16(N), K-step 32.
// ---------------------------------------------------------------------------
enum { EPI_NONE = 0, EPI_BIAS = 1, EPI_RELU_GATE = 2 };

template <int MODE, bool OUTBF16>
__global__ __launch_bounds__(256) void gemm_wmma_kernel(
    const unsigned short* __restrict__ A, int lda, long aBatch,
    const unsigned short* __restrict__ Bm, int ldb, long bBatch,
    void* __restrict__ Cp, int ldc, long cBatch,
    const float* __restrict__ bias, long biasBatch,
    const float* __restrict__ gate, int gateStride,
    float scale, int M, int N, int K) {
  __shared__ unsigned short As[64 * 32];
  __shared__ unsigned short BsT[64 * 32];   // [n][k] transposed
  int tid = threadIdx.x;
  int wave = tid >> 5, lane = tid & 31;
  int half = lane >> 4, ln = lane & 15;
  int z = blockIdx.z;
  int m0 = blockIdx.x * 64;
  int n0 = blockIdx.y * 64;
  int wm = (wave & 1) * 32;
  int wn = (wave >> 1) * 16;

  const unsigned short* Ab = A + (size_t)z * aBatch;
  const unsigned short* Bb = Bm + (size_t)z * bBatch;

  v8f acc[2] = {v8f_zero(), v8f_zero()};

  int ar = tid >> 2;            // 0..63
  int ac = (tid & 3) * 8;       // 0,8,16,24
  int br = tid >> 3;            // 0..31
  int bc = (tid & 7) * 8;       // 0..56

  for (int k0 = 0; k0 < K; k0 += 32) {
    // A tile 64x32, vectorized 16B per thread
    const uint4* ap =
        (const uint4*)(Ab + (size_t)(m0 + ar) * lda + (k0 + ac));
    *(uint4*)(&As[ar * 32 + ac]) = *ap;
    if (k0 + 32 < K)
      __builtin_prefetch(Ab + (size_t)(m0 + ar) * lda + (k0 + 32 + ac), 0, 1);
    // B tile 32x64, transposed into LDS with N-tail guard
#pragma unroll
    for (int j = 0; j < 8; ++j) {
      int nn = n0 + bc + j;
      unsigned short bv = 0;
      if (nn < N) bv = Bb[(size_t)(k0 + br) * ldb + nn];
      BsT[(bc + j) * 32 + br] = bv;
    }
    __syncthreads();

    Frag fb;
#pragma unroll
    for (int i = 0; i < 8; ++i) {
      int kk = (i & 3) * 2 + (i >> 2) * 16 + half * 8;
      fb.u[i] = *(const unsigned int*)(&BsT[(wn + ln) * 32 + kk]);
    }
    Frag fa0, fa1;
#pragma unroll
    for (int i = 0; i < 8; ++i) {
      int kb = (i & 3) * 2 + (i >> 2) * 16 + half * 8;
      fa0.u[i] = *(const unsigned int*)(&As[(wm + ln) * 32 + kb]);
      fa1.u[i] = *(const unsigned int*)(&As[(wm + 16 + ln) * 32 + kb]);
    }
    acc[0] = wmma_bf16(fa0, fb, acc[0]);
    acc[1] = wmma_bf16(fa1, fb, acc[1]);
    __syncthreads();
  }

  int nc = n0 + wn + ln;                      // column within N
  bool nok = nc < N;
  float bval = 0.f;
  if (MODE != EPI_NONE)
    bval = nok ? bias[(size_t)z * biasBatch + nc] : 0.f;

#pragma unroll
  for (int t2 = 0; t2 < 2; ++t2) {
#pragma unroll
    for (int r = 0; r < 8; ++r) {
      int row = m0 + wm + t2 * 16 + r + half * 8;
      float v = acc[t2][r];
      if (MODE == EPI_BIAS) v = (v + bval) * scale;
      if (MODE == EPI_RELU_GATE) {
        v = v + bval;
        v = v > 0.f ? v : 0.f;
        v *= gate[(size_t)row * gateStride + z];
      }
      if (nok) {
        size_t idx = (size_t)row * ldc + (size_t)z * cBatch + nc;
        if (OUTBF16) ((unsigned short*)Cp)[idx] = f2bf(v);
        else ((float*)Cp)[idx] = v;
      }
    }
  }
}

// ---------------------------------------------------------------------------
// Fused flash attention, one (b, expert) per blockIdx.{z,y}, 128 q-rows/block.
// Q pre-scaled bf16 [T, EA*HD]; K/V bf16 [T, HD]; rel f32 [T, EA*129];
// gates f32 [T, EA]; Ctx bf16 [T, EA*HD] (gate-scaled).
// ---------------------------------------------------------------------------
__global__ __launch_bounds__(256) void moa_flash_kernel(
    const unsigned short* __restrict__ Q, const unsigned short* __restrict__ Kb,
    const unsigned short* __restrict__ Vb, const float* __restrict__ rel,
    const float* __restrict__ gates, unsigned short* __restrict__ Ctx) {
  const int S = 1024, EA = 8, R129 = 129, RLD = EA * R129;
  __shared__ unsigned short Ks[32 * 128];      // [key][h]
  __shared__ unsigned short Vt[128 * 34];      // [h][key], padded stride
  __shared__ unsigned short Ps[8 * 16 * 32];   // per-wave P tile [m][key]

  int b = blockIdx.z, e = blockIdx.y, q0 = blockIdx.x * 128;
  int tid = threadIdx.x, wave = tid >> 5, lane = tid & 31;
  int half = lane >> 4, ln = lane & 15;
  int qw = q0 + wave * 16;
  size_t tokQ = (size_t)b * S + qw;

  // Q A-fragments (4 K-steps of 32 over HD=128)
  Frag fq[4];
#pragma unroll
  for (int f = 0; f < 4; ++f)
#pragma unroll
    for (int i = 0; i < 8; ++i) {
      int kb = (i & 3) * 2 + (i >> 2) * 16 + half * 8 + f * 32;
      fq[f].u[i] =
          *(const unsigned int*)(Q + (tokQ + ln) * 1024 + e * 128 + kb);
    }

  float mrow[8], lrow[8];
  v8f o[8];
#pragma unroll
  for (int r = 0; r < 8; ++r) { mrow[r] = -1e30f; lrow[r] = 0.f; }
#pragma unroll
  for (int j = 0; j < 8; ++j) o[j] = v8f_zero();

  int lrw = tid >> 3;            // 0..31 key row for staging
  int lc0 = (tid & 7) * 16;      // 16 cols

  for (int kt = 0; kt < 32; ++kt) {
    size_t kbase = ((size_t)b * S + kt * 32 + lrw) * 128 + lc0;
    *(uint4*)(&Ks[lrw * 128 + lc0])     = *(const uint4*)(Kb + kbase);
    *(uint4*)(&Ks[lrw * 128 + lc0 + 8]) = *(const uint4*)(Kb + kbase + 8);
    unsigned short vt[16];
    *(uint4*)(&vt[0]) = *(const uint4*)(Vb + kbase);
    *(uint4*)(&vt[8]) = *(const uint4*)(Vb + kbase + 8);
#pragma unroll
    for (int j = 0; j < 16; ++j) Vt[(lc0 + j) * 34 + lrw] = vt[j];
    __syncthreads();

    // scores: two 16-key sub-tiles
    v8f s0 = v8f_zero(), s1 = v8f_zero();
#pragma unroll
    for (int f = 0; f < 4; ++f) {
      Frag kb0, kb1;
#pragma unroll
      for (int i = 0; i < 8; ++i) {
        int hh = (i & 3) * 2 + (i >> 2) * 16 + half * 8 + f * 32;
        kb0.u[i] = *(const unsigned int*)(&Ks[ln * 128 + hh]);
        kb1.u[i] = *(const unsigned int*)(&Ks[(16 + ln) * 128 + hh]);
      }
      s0 = wmma_bf16(fq[f], kb0, s0);
      s1 = wmma_bf16(fq[f], kb1, s1);
    }

    // Shaw relative-position bias (precomputed rel_s gather)
#pragma unroll
    for (int r = 0; r < 8; ++r) {
      int qr = qw + r + half * 8;
      size_t rbase = ((size_t)b * S + qr) * RLD + (size_t)e * R129;
      int kc0 = kt * 32 + ln;
      int d0 = kc0 - qr; d0 = d0 < -64 ? -64 : (d0 > 64 ? 64 : d0);
      int kc1 = kc0 + 16;
      int d1 = kc1 - qr; d1 = d1 < -64 ? -64 : (d1 > 64 ? 64 : d1);
      s0[r] += rel[rbase + d0 + 64];
      s1[r] += rel[rbase + d1 + 64];
    }

    // online softmax (row reductions across 16 lanes within half-group)
#pragma unroll
    for (int r = 0; r < 8; ++r) {
      float mx = fmaxf(s0[r], s1[r]);
      mx = fmaxf(mx, __shfl_xor(mx, 1, 32));
      mx = fmaxf(mx, __shfl_xor(mx, 2, 32));
      mx = fmaxf(mx, __shfl_xor(mx, 4, 32));
      mx = fmaxf(mx, __shfl_xor(mx, 8, 32));
      float mnew = fmaxf(mrow[r], mx);
      float fac = __expf(mrow[r] - mnew);
      mrow[r] = mnew;
      float p0 = __expf(s0[r] - mnew);
      float p1 = __expf(s1[r] - mnew);
      s0[r] = p0; s1[r] = p1;
      float sum = p0 + p1;
      sum += __shfl_xor(sum, 1, 32);
      sum += __shfl_xor(sum, 2, 32);
      sum += __shfl_xor(sum, 4, 32);
      sum += __shfl_xor(sum, 8, 32);
      lrow[r] = lrow[r] * fac + sum;
#pragma unroll
      for (int j = 0; j < 8; ++j) o[j][r] *= fac;
    }

    // repack P (C-layout) -> A-fragment via wave-private LDS
    unsigned short* pw = &Ps[wave * 512];
#pragma unroll
    for (int r = 0; r < 8; ++r) {
      int m = r + half * 8;
      pw[m * 32 + ln]      = f2bf(s0[r]);
      pw[m * 32 + 16 + ln] = f2bf(s1[r]);
    }
    Frag fp;
#pragma unroll
    for (int i = 0; i < 8; ++i) {
      int kb = (i & 3) * 2 + (i >> 2) * 16 + half * 8;
      fp.u[i] = *(const unsigned int*)(&pw[ln * 32 + kb]);
    }

    // O += P @ V  (8 HD sub-tiles of 16)
#pragma unroll
    for (int j = 0; j < 8; ++j) {
      Frag fv;
#pragma unroll
      for (int i = 0; i < 8; ++i) {
        int kk = (i & 3) * 2 + (i >> 2) * 16 + half * 8;
        fv.u[i] = *(const unsigned int*)(&Vt[(j * 16 + ln) * 34 + kk]);
      }
      o[j] = wmma_bf16(fp, fv, o[j]);
    }
    __syncthreads();
  }

  // epilogue: divide by l, scale by router gate, store bf16 ctx'
  float gr[8];
#pragma unroll
  for (int r = 0; r < 8; ++r) {
    int qr = qw + r + half * 8;
    gr[r] = gates[((size_t)b * S + qr) * EA + e] / lrow[r];
  }
#pragma unroll
  for (int j = 0; j < 8; ++j)
#pragma unroll
    for (int r = 0; r < 8; ++r) {
      int qr = qw + r + half * 8;
      Ctx[((size_t)b * S + qr) * 1024 + e * 128 + j * 16 + ln] =
          f2bf(o[j][r] * gr[r]);
    }
}

// ---------------------------------------------------------------------------
// Y[i] = Y[i] + Res[i] + sum_e Gates[row,e] * Ob[e,d]
// ---------------------------------------------------------------------------
template <int E>
__global__ void residual_gatebias_kernel(float* __restrict__ Y,
                                         const float* __restrict__ Res,
                                         const float* __restrict__ Gates,
                                         const float* __restrict__ Ob, int D,
                                         long total) {
  long i = (long)blockIdx.x * blockDim.x + threadIdx.x;
  if (i >= total) return;
  long row = i / D;
  int d = (int)(i - row * D);
  float acc = Y[i] + Res[i];
#pragma unroll
  for (int e = 0; e < E; ++e)
    acc += Gates[row * E + e] * Ob[(size_t)e * D + d];
  Y[i] = acc;
}

// ---------------------------------------------------------------------------
// host launch
// ---------------------------------------------------------------------------
extern "C" void kernel_launch(void* const* d_in, const int* in_sizes, int n_in,
                              void* d_out, int out_size, void* d_ws,
                              size_t ws_size, hipStream_t stream) {
  (void)in_sizes; (void)n_in; (void)out_size; (void)ws_size;
  const int Bn = 4, S = 1024, T = Bn * S, D = 1024, HD = 128;
  const int EA = 8, EF = 16, FH = 512;

  const float* src    = (const float*)d_in[0];
  const float* ln1_w  = (const float*)d_in[1];
  const float* ln1_b  = (const float*)d_in[2];
  const float* ln2_w  = (const float*)d_in[3];
  const float* ln2_b  = (const float*)d_in[4];
  const float* agw    = (const float*)d_in[5];
  const float* q_w    = (const float*)d_in[6];
  const float* q_b    = (const float*)d_in[7];
  const float* k_w    = (const float*)d_in[8];
  const float* k_b    = (const float*)d_in[9];
  const float* v_w    = (const float*)d_in[10];
  const float* v_b    = (const float*)d_in[11];
  const float* o_w    = (const float*)d_in[12];
  const float* o_b    = (const float*)d_in[13];
  const float* relt   = (const float*)d_in[14];
  const float* fgw    = (const float*)d_in[15];
  const float* ff_w1  = (const float*)d_in[16];
  const float* ff_b1  = (const float*)d_in[17];
  const float* ff_w2  = (const float*)d_in[18];
  const float* ff_b2  = (const float*)d_in[19];
  float* outp = (float*)d_out;

  char* ws = (char*)d_ws;
  size_t off = 0;
  auto alloc = [&](size_t bytes) -> void* {
    void* p = ws + off;
    off = (off + bytes + 255) & ~(size_t)255;
    return p;
  };
  typedef unsigned short u16;
  u16*   xn1  = (u16*)alloc((size_t)T * D * 2);
  u16*   qbf  = (u16*)alloc((size_t)T * EA * HD * 2);
  u16*   kbf  = (u16*)alloc((size_t)T * HD * 2);
  u16*   vbf  = (u16*)alloc((size_t)T * HD * 2);
  float* rels = (float*)alloc((size_t)T * EA * 129 * 4);
  float* ga   = (float*)alloc((size_t)T * EA * 4);
  u16*   ctx  = (u16*)alloc((size_t)T * EA * HD * 2);
  float* xbuf = (float*)alloc((size_t)T * D * 4);
  u16*   xn2  = (u16*)alloc((size_t)T * D * 2);
  float* gf   = (float*)alloc((size_t)T * EF * 4);
  u16*   hp   = (u16*)alloc((size_t)T * EF * FH * 2);
  u16*   wq   = (u16*)alloc((size_t)EA * D * HD * 2);
  u16*   wk   = (u16*)alloc((size_t)D * HD * 2);
  u16*   wv   = (u16*)alloc((size_t)D * HD * 2);
  u16*   wo   = (u16*)alloc((size_t)EA * HD * D * 2);
  u16*   w1   = (u16*)alloc((size_t)EF * D * FH * 2);
  u16*   w2   = (u16*)alloc((size_t)EF * FH * D * 2);
  u16*   relT = (u16*)alloc((size_t)128 * 129 * 2);

  auto cvt = [&](const float* in, u16* out, long n) {
    to_bf16_kernel<<<(int)((n + 255) / 256), 256, 0, stream>>>(in, out, n);
  };
  cvt(q_w, wq, (long)EA * D * HD);
  cvt(k_w, wk, (long)D * HD);
  cvt(v_w, wv, (long)D * HD);
  cvt(o_w, wo, (long)EA * HD * D);
  cvt(ff_w1, w1, (long)EF * D * FH);
  cvt(ff_w2, w2, (long)EF * FH * D);
  transpose_rel_kernel<<<(128 * 129 + 255) / 256, 256, 0, stream>>>(relt, relT);

  // LN1 + attention router
  layernorm_bf16_kernel<<<T, 256, 0, stream>>>(src, ln1_w, ln1_b, xn1, D);
  router_top2_kernel<8><<<T, 256, 0, stream>>>(xn1, agw, ga, D);

  const float qscale = 0.08838834764831845f;  // 128^-0.5
  // q (batched over 8 experts), bf16 out, scale folded
  gemm_wmma_kernel<EPI_BIAS, true><<<dim3(T / 64, HD / 64, EA), 256, 0, stream>>>(
      xn1, D, 0L, wq, HD, (long)D * HD, qbf, EA * HD, (long)HD,
      q_b, (long)HD, nullptr, 0, qscale, T, HD, D);
  // k, v
  gemm_wmma_kernel<EPI_BIAS, true><<<dim3(T / 64, HD / 64, 1), 256, 0, stream>>>(
      xn1, D, 0L, wk, HD, 0L, kbf, HD, 0L, k_b, 0L, nullptr, 0, 1.f, T, HD, D);
  gemm_wmma_kernel<EPI_BIAS, true><<<dim3(T / 64, HD / 64, 1), 256, 0, stream>>>(
      xn1, D, 0L, wv, HD, 0L, vbf, HD, 0L, v_b, 0L, nullptr, 0, 1.f, T, HD, D);
  // rel_s = q @ rel_table^T : [T, EA*129], N=129 with tail guards
  gemm_wmma_kernel<EPI_NONE, false><<<dim3(T / 64, 3, EA), 256, 0, stream>>>(
      qbf, EA * HD, (long)HD, relT, 129, 0L, rels, EA * 129, 129L,
      nullptr, 0L, nullptr, 0, 1.f, T, 129, HD);

  // fused flash attention -> gate-scaled ctx'
  moa_flash_kernel<<<dim3(S / 128, EA, Bn), 256, 0, stream>>>(qbf, kbf, vbf,
                                                              rels, ga, ctx);
  // output combine over concatenated o_w  [EA*HD, D]
  gemm_wmma_kernel<EPI_NONE, false><<<dim3(T / 64, D / 64, 1), 256, 0, stream>>>(
      ctx, EA * HD, 0L, wo, D, 0L, xbuf, D, 0L,
      nullptr, 0L, nullptr, 0, 1.f, T, D, EA * HD);
  // x = src + attn + gates @ o_b
  residual_gatebias_kernel<8><<<(int)(((long)T * D + 255) / 256), 256, 0,
                                stream>>>(xbuf, src, ga, o_b, D, (long)T * D);

  // LN2 + FFN router
  layernorm_bf16_kernel<<<T, 256, 0, stream>>>(xbuf, ln2_w, ln2_b, xn2, D);
  router_top2_kernel<16><<<T, 256, 0, stream>>>(xn2, fgw, gf, D);

  // h' = gate * relu(xn2 @ W1 + b1), bf16 [T, EF*FH]
  gemm_wmma_kernel<EPI_RELU_GATE, true><<<dim3(T / 64, FH / 64, EF), 256, 0,
                                          stream>>>(
      xn2, D, 0L, w1, FH, (long)D * FH, hp, EF * FH, (long)FH,
      ff_b1, (long)FH, gf, EF, 1.f, T, FH, D);
  // out = h' @ W2_concat [EF*FH, D] -> d_out
  gemm_wmma_kernel<EPI_NONE, false><<<dim3(T / 64, D / 64, 1), 256, 0, stream>>>(
      hp, EF * FH, 0L, w2, D, 0L, outp, D, 0L,
      nullptr, 0L, nullptr, 0, 1.f, T, D, EF * FH);
  // final residual: out += x + gates_f @ ff_b2
  residual_gatebias_kernel<16><<<(int)(((long)T * D + 255) / 256), 256, 0,
                                 stream>>>(outp, xbuf, gf, ff_b2, D,
                                           (long)T * D);
}
